// SpatialClassifier_11940009083650
// MI455X (gfx1250) — compile-verified, matrix operand order checked by reading, not run on hardware
//
#include <hip/hip_runtime.h>
#include <hip/hip_bf16.h>

typedef __attribute__((ext_vector_type(16))) _Float16 v16h;
typedef __attribute__((ext_vector_type(8)))  float    v8f;

#define NQ   8192
#define NC   16384
#define HDIM 256
#define FDIM 128
#define KNN  32

// ws layout (bytes)
#define NBR_OFF   0u
#define Y_OFF     (1u << 20)            // 8192*32*4 = 1MB for nbr
#define FRAG_OFF  (5u << 20)            // y is 8192*128*4 = 4MB

// fragment ids (each frag = 512 halves = 1KB)
#define FR_NN1   0
#define FR_NN2   32
#define FR_LIN1  64
#define FR_LIN2  128
#define FR_CLS1  160
#define FR_CLS2  192
#define FR_PROP1 196
#define FR_PROP2 228

typedef const __attribute__((address_space(1))) v16h* gv16h_ptr;

__device__ __forceinline__ float ssp_f(float x) {
    float sp = (x > 15.f) ? x : log1pf(__expf(x));
    return sp - 0.69314718056f;
}

// A-fragment (16x32 f16) from row-major f16 LDS tile, per CDNA5 16-bit A layout:
// lanes 0-15: M=lane, VGPR v<4 -> K=2v+c, v>=4 -> K=16+2(v-4)+c ; lanes 16-31: +8 on K-halves
__device__ __forceinline__ v16h load_a_frag(const _Float16* base, int ld, int kBase, int lane) {
    int m = lane & 15;
    int hioff = (lane & 16) ? 8 : 0;
    v16h a;
#pragma unroll
    for (int h = 0; h < 16; ++h) {
        int vv = h >> 1, cc = h & 1;
        int k = (vv < 4) ? (2 * vv + cc + hioff) : (16 + 2 * (vv - 4) + cc + hioff);
        a[h] = base[m * ld + kBase + k];
    }
    return a;
}

// B-fragment: pre-swizzled in ws, 32B contiguous per lane -> 2x global_load_b128.
// The empty asm makes the address opaque so LICM cannot hoist (and then spill)
// all fragment loads out of the GEMM loops; the address_space(1) cast restores
// global provenance so the load lowers to global_load (LOADcnt only), not
// flat_load (which would also tie up DScnt against the LDS A-fragment stream).
__device__ __forceinline__ v16h load_b_frag(const _Float16* frags, int fragId, int lane) {
    const _Float16* p = frags + (size_t)fragId * 512 + lane * 16;
    asm volatile("" : "+v"(p));
    return *(gv16h_ptr)(uintptr_t)p;
}

__device__ __forceinline__ v8f wmma16(v16h a, v16h b, v8f c) {
    return __builtin_amdgcn_wmma_f32_16x16x32_f16(false, a, false, b, (short)0, c, false, false);
}

// ---------------- weight prep: f32 [fin][foutReal] -> f16 fragment-major ----------------
__global__ void prep_weight_kernel(const float* __restrict__ W, int fin, int foutPad, int foutReal,
                                   _Float16* __restrict__ dst) {
    int t = blockIdx.x * blockDim.x + threadIdx.x;
    int lane = t & 31;
    int frag = t >> 5;
    int ntiles = foutPad >> 4;
    int nfrags = (fin >> 5) * ntiles;
    if (frag >= nfrags) return;
    int kc = frag / ntiles, nt = frag % ntiles;
    int n = nt * 16 + (lane & 15);
    int hioff = (lane & 16) ? 8 : 0;
    _Float16* out = dst + (size_t)frag * 512 + lane * 16;
#pragma unroll
    for (int h = 0; h < 16; ++h) {
        int vv = h >> 1, cc = h & 1;
        int k = (vv < 4) ? (2 * vv + cc + hioff) : (16 + 2 * (vv - 4) + cc + hioff);
        int kk = kc * 32 + k;
        float val = (n < foutReal) ? W[(size_t)kk * foutReal + n] : 0.f;
        out[h] = (_Float16)val;
    }
}

// ---------------- exact batch-aware kNN: one wave32 per query ----------------
__global__ __launch_bounds__(128) void knn_kernel(const float* __restrict__ posq,
                                                  const float* __restrict__ posc,
                                                  const int* __restrict__ bq,
                                                  const int* __restrict__ bc,
                                                  int* __restrict__ nbr) {
    int w = threadIdx.x >> 5;
    int lane = threadIdx.x & 31;
    int q = blockIdx.x * 4 + w;
    float qx = posq[q * 3 + 0], qy = posq[q * 3 + 1], qz = posq[q * 3 + 2];
    int qb = bq[q];

    float bd[32];
    int bi[32];
#pragma unroll
    for (int p = 0; p < 32; ++p) { bd[p] = 3.4e38f; bi[p] = 0; }

    for (int c = lane; c < NC; c += 32) {
        if (bc[c] != qb) continue;
        float dx = qx - posc[c * 3 + 0];
        float dy = qy - posc[c * 3 + 1];
        float dz = qz - posc[c * 3 + 2];
        float d2 = dx * dx + dy * dy + dz * dz;
        if (d2 < bd[31]) {
            float cur = d2; int curi = c;
#pragma unroll
            for (int p = 0; p < 32; ++p) {
                if (cur < bd[p]) {
                    float tf = bd[p]; bd[p] = cur; cur = tf;
                    int ti = bi[p]; bi[p] = curi; curi = ti;
                }
            }
        }
    }
    // merge 32 sorted per-lane lists: 32 rounds of wave-min
    for (int r = 0; r < KNN; ++r) {
        float v = bd[0]; int ii = bi[0]; int wl = lane;
#pragma unroll
        for (int off = 16; off > 0; off >>= 1) {
            float ov = __shfl_xor(v, off, 32);
            int   oi = __shfl_xor(ii, off, 32);
            int   ow = __shfl_xor(wl, off, 32);
            if (ov < v || (ov == v && ow < wl)) { v = ov; ii = oi; wl = ow; }
        }
        if (lane == 0) nbr[q * KNN + r] = ii;
        if (lane == wl) {  // pop head
#pragma unroll
            for (int p = 0; p < 31; ++p) { bd[p] = bd[p + 1]; bi[p] = bi[p + 1]; }
            bd[31] = 3.4e38f; bi[31] = 0;
        }
    }
}

// ---------------- edge filter network: one wave per query (2 tiles of 16 edges) ----------------
__global__ __launch_bounds__(128) __attribute__((amdgpu_waves_per_eu(2)))
void edge_kernel(const float* __restrict__ posq,
                 const float* __restrict__ posc,
                 const float* __restrict__ attr,
                 const int* __restrict__ nbr,
                 const _Float16* __restrict__ frags,
                 const float* __restrict__ nn1_b,
                 const float* __restrict__ nn2_b,
                 const float* __restrict__ lin2_b,
                 float* __restrict__ y) {
    __shared__ _Float16 ldsA[4][16][HDIM];
    __shared__ float sdist[4][16];
    __shared__ int   sj[4][16];

    int w = threadIdx.x >> 5;
    int lane = threadIdx.x & 31;
    int q = blockIdx.x * 4 + w;
    int n0 = lane & 15;
    int mo = (lane & 16) ? 8 : 0;

    float qx = posq[q * 3 + 0], qy = posq[q * 3 + 1], qz = posq[q * 3 + 2];

    float ycol[8];
#pragma unroll
    for (int n = 0; n < 8; ++n) ycol[n] = 0.f;

    const float DELTA = 10.f / 127.f;
    const float COEF  = -0.5f / (DELTA * DELTA);

    for (int t = 0; t < 2; ++t) {
        __syncthreads();
        if (lane < 16) {
            int e = q * KNN + t * 16 + lane;
            int j = nbr[e];
            float dx = qx - posc[j * 3 + 0];
            float dy = qy - posc[j * 3 + 1];
            float dz = qz - posc[j * 3 + 2];
            sdist[w][lane] = sqrtf(dx * dx + dy * dy + dz * dz);
            sj[w][lane] = j;
        }
        __syncthreads();
        // Gaussian smearing -> ldsA[16][0..127]
        for (int idx = lane; idx < 16 * FDIM; idx += 32) {
            int row = idx >> 7, f = idx & 127;
            float dd = sdist[w][row] - (float)f * DELTA;
            ldsA[w][row][f] = (_Float16)__expf(COEF * dd * dd);
        }
        __syncthreads();

        // GEMM1: t1 = g @ nn1 + b1
        v8f acc[8];
#pragma unroll
        for (int n = 0; n < 8; ++n) {
            float bv = nn1_b[n * 16 + n0];
#pragma unroll
            for (int r = 0; r < 8; ++r) acc[n][r] = bv;
        }
#pragma unroll
        for (int kc = 0; kc < 4; ++kc) {
            v16h a = load_a_frag(&ldsA[w][0][0], HDIM, kc * 32, lane);
#pragma unroll
            for (int n = 0; n < 8; ++n)
                acc[n] = wmma16(a, load_b_frag(frags, FR_NN1 + kc * 8 + n, lane), acc[n]);
        }
        // ssp, write back as f16
#pragma unroll
        for (int n = 0; n < 8; ++n)
#pragma unroll
            for (int r = 0; r < 8; ++r)
                ldsA[w][mo + r][n * 16 + n0] = (_Float16)ssp_f(acc[n][r]);
        __syncthreads();

        // GEMM2: W = t1 @ nn2 + b2
        v8f Wacc[8];
#pragma unroll
        for (int n = 0; n < 8; ++n) {
            float bv = nn2_b[n * 16 + n0];
#pragma unroll
            for (int r = 0; r < 8; ++r) Wacc[n][r] = bv;
        }
#pragma unroll
        for (int kc = 0; kc < 4; ++kc) {
            v16h a = load_a_frag(&ldsA[w][0][0], HDIM, kc * 32, lane);
#pragma unroll
            for (int n = 0; n < 8; ++n)
                Wacc[n] = wmma16(a, load_b_frag(frags, FR_NN2 + kc * 8 + n, lane), Wacc[n]);
        }
        __syncthreads();

        // stage attr_j [16 x 256] as f16 (float4 gathers)
        for (int idx = lane; idx < 16 * 64; idx += 32) {
            int row = idx >> 6, c4 = (idx & 63) * 4;
            int j = sj[w][row];
            float4 v4 = *reinterpret_cast<const float4*>(attr + (size_t)j * HDIM + c4);
            ldsA[w][row][c4 + 0] = (_Float16)v4.x;
            ldsA[w][row][c4 + 1] = (_Float16)v4.y;
            ldsA[w][row][c4 + 2] = (_Float16)v4.z;
            ldsA[w][row][c4 + 3] = (_Float16)v4.w;
        }
        __syncthreads();

        // GEMM3: s = attr_j @ lin1 (no bias)
        v8f sacc[8];
#pragma unroll
        for (int n = 0; n < 8; ++n)
#pragma unroll
            for (int r = 0; r < 8; ++r) sacc[n][r] = 0.f;
#pragma unroll
        for (int kc = 0; kc < 8; ++kc) {
            v16h a = load_a_frag(&ldsA[w][0][0], HDIM, kc * 32, lane);
#pragma unroll
            for (int n = 0; n < 8; ++n)
                sacc[n] = wmma16(a, load_b_frag(frags, FR_LIN1 + kc * 8 + n, lane), sacc[n]);
        }
        __syncthreads();
        // Hadamard W*s -> f16
#pragma unroll
        for (int n = 0; n < 8; ++n)
#pragma unroll
            for (int r = 0; r < 8; ++r)
                ldsA[w][mo + r][n * 16 + n0] = (_Float16)(Wacc[n][r] * sacc[n][r]);
        __syncthreads();

        // GEMM4: h = (W*s) @ lin2 + b
        v8f hacc[8];
#pragma unroll
        for (int n = 0; n < 8; ++n) {
            float bv = lin2_b[n * 16 + n0];
#pragma unroll
            for (int r = 0; r < 8; ++r) hacc[n][r] = bv;
        }
#pragma unroll
        for (int kc = 0; kc < 4; ++kc) {
            v16h a = load_a_frag(&ldsA[w][0][0], HDIM, kc * 32, lane);
#pragma unroll
            for (int n = 0; n < 8; ++n)
                hacc[n] = wmma16(a, load_b_frag(frags, FR_LIN2 + kc * 8 + n, lane), hacc[n]);
        }
        // cosine cutoff + row-reduce into ycol
#pragma unroll
        for (int r = 0; r < 8; ++r) {
            float d = sdist[w][mo + r];
            float Ce = (d <= 10.f) ? 0.5f * (__cosf(d * 0.31415926535f) + 1.f) : 0.f;
#pragma unroll
            for (int n = 0; n < 8; ++n) ycol[n] += hacc[n][r] * Ce;
        }
    }
    // combine M 0-7 (lanes 0-15) with M 8-15 (lanes 16-31)
#pragma unroll
    for (int n = 0; n < 8; ++n) ycol[n] += __shfl_xor(ycol[n], 16, 32);
    if (lane < 16) {
#pragma unroll
        for (int n = 0; n < 8; ++n) y[(size_t)q * FDIM + n * 16 + lane] = ycol[n];
    }
}

// ---------------- output heads: one wave per 16 queries ----------------
__global__ __launch_bounds__(128) __attribute__((amdgpu_waves_per_eu(2)))
void head_kernel(const float* __restrict__ y,
                 const _Float16* __restrict__ frags,
                 const float* __restrict__ cls1_b,
                 const float* __restrict__ cls2_b,
                 const float* __restrict__ prop1_b,
                 const float* __restrict__ prop2_b,
                 float* __restrict__ out) {
    __shared__ _Float16 ldsY[4][16][FDIM];
    __shared__ _Float16 ldsT[4][16][FDIM];
    int w = threadIdx.x >> 5;
    int lane = threadIdx.x & 31;
    int q0 = (blockIdx.x * 4 + w) * 16;
    int n0 = lane & 15;
    int mo = (lane & 16) ? 8 : 0;

    for (int idx = lane; idx < 16 * FDIM; idx += 32) {
        int row = idx >> 7, f = idx & 127;
        ldsY[w][row][f] = (_Float16)y[(size_t)(q0 + row) * FDIM + f];
    }
    __syncthreads();

    float* out_cls = out;
    float* out_ind = out + (size_t)NQ * 7;

    for (int head = 0; head < 2; ++head) {
        int fr1 = head ? FR_PROP1 : FR_CLS1;
        int fr2 = head ? FR_PROP2 : FR_CLS2;
        const float* b1 = head ? prop1_b : cls1_b;
        const float* b2 = head ? prop2_b : cls2_b;
        int nout = head ? 8 : 7;
        float* dst = head ? out_ind : out_cls;

        v8f acc[8];
#pragma unroll
        for (int n = 0; n < 8; ++n) {
            float bv = b1[n * 16 + n0];
#pragma unroll
            for (int r = 0; r < 8; ++r) acc[n][r] = bv;
        }
#pragma unroll
        for (int kc = 0; kc < 4; ++kc) {
            v16h a = load_a_frag(&ldsY[w][0][0], FDIM, kc * 32, lane);
#pragma unroll
            for (int n = 0; n < 8; ++n)
                acc[n] = wmma16(a, load_b_frag(frags, fr1 + kc * 8 + n, lane), acc[n]);
        }
#pragma unroll
        for (int n = 0; n < 8; ++n)
#pragma unroll
            for (int r = 0; r < 8; ++r)
                ldsT[w][mo + r][n * 16 + n0] = (_Float16)ssp_f(acc[n][r]);
        __syncthreads();

        v8f o;
        {
            float bv = (n0 < nout) ? b2[n0] : 0.f;
#pragma unroll
            for (int r = 0; r < 8; ++r) o[r] = bv;
        }
#pragma unroll
        for (int kc = 0; kc < 4; ++kc) {
            v16h a = load_a_frag(&ldsT[w][0][0], FDIM, kc * 32, lane);
            o = wmma16(a, load_b_frag(frags, fr2 + kc, lane), o);
        }
        if (n0 < nout) {
#pragma unroll
            for (int r = 0; r < 8; ++r)
                dst[(size_t)(q0 + mo + r) * nout + n0] = o[r];
        }
        __syncthreads();
    }
}

extern "C" void kernel_launch(void* const* d_in, const int* in_sizes, int n_in,
                              void* d_out, int out_size, void* d_ws, size_t ws_size,
                              hipStream_t stream) {
    (void)in_sizes; (void)n_in; (void)out_size; (void)ws_size;
    const float* pos_q  = (const float*)d_in[0];
    const float* pos_c  = (const float*)d_in[1];
    const float* attr   = (const float*)d_in[2];
    const int*   bq     = (const int*)d_in[3];
    const int*   bc     = (const int*)d_in[4];
    const float* lin1_w = (const float*)d_in[5];
    const float* lin2_w = (const float*)d_in[6];
    const float* lin2_b = (const float*)d_in[7];
    const float* nn1_w  = (const float*)d_in[8];
    const float* nn1_b  = (const float*)d_in[9];
    const float* nn2_w  = (const float*)d_in[10];
    const float* nn2_b  = (const float*)d_in[11];
    const float* cls1_w = (const float*)d_in[12];
    const float* cls1_b = (const float*)d_in[13];
    const float* cls2_w = (const float*)d_in[14];
    const float* cls2_b = (const float*)d_in[15];
    const float* prop1_w = (const float*)d_in[16];
    const float* prop1_b = (const float*)d_in[17];
    const float* prop2_w = (const float*)d_in[18];
    const float* prop2_b = (const float*)d_in[19];

    int*       nbr   = (int*)((char*)d_ws + NBR_OFF);
    float*     y     = (float*)((char*)d_ws + Y_OFF);
    _Float16*  frags = (_Float16*)((char*)d_ws + FRAG_OFF);

    auto prep = [&](const float* W, int fin, int foutPad, int foutReal, int fragBase) {
        int nf = (fin / 32) * (foutPad / 16);
        int threads = nf * 32;
        prep_weight_kernel<<<(threads + 127) / 128, 128, 0, stream>>>(
            W, fin, foutPad, foutReal, frags + (size_t)fragBase * 512);
    };
    prep(nn1_w,  128, 128, 128, FR_NN1);
    prep(nn2_w,  128, 128, 128, FR_NN2);
    prep(lin1_w, 256, 128, 128, FR_LIN1);
    prep(lin2_w, 128, 128, 128, FR_LIN2);
    prep(cls1_w, 128, 128, 128, FR_CLS1);
    prep(cls2_w, 128,  16,   7, FR_CLS2);
    prep(prop1_w,128, 128, 128, FR_PROP1);
    prep(prop2_w,128,  16,   8, FR_PROP2);

    knn_kernel<<<NQ / 4, 128, 0, stream>>>(pos_q, pos_c, bq, bc, nbr);
    edge_kernel<<<NQ / 4, 128, 0, stream>>>(pos_q, pos_c, attr, nbr, frags,
                                            nn1_b, nn2_b, lin2_b, y);
    head_kernel<<<NQ / 16 / 4, 128, 0, stream>>>(y, frags, cls1_b, cls2_b,
                                                 prop1_b, prop2_b, (float*)d_out);
}
